// SolvGNNV6_37778532335676
// MI455X (gfx1250) — compile-verified, matrix operand order checked by reading, not run on hardware
//
#include <hip/hip_runtime.h>
#include <hip/hip_bf16.h>

// ---------------------------------------------------------------------------
// Problem constants (match reference)
// ---------------------------------------------------------------------------
#define NB     2048          // graphs per side
#define NPG    40            // nodes per graph
#define NN     (NB * NPG)    // 81920 nodes
#define NE     (4 * NN)      // 327680 edges
#define IN_DIM 74
#define K1PAD  96            // 74 padded to multiple of 32
#define HDIM   256
#define ADDF   8
#define GCRL   5
#define M1     1024
#define M2     512
#define ZDIM   544           // (2H + ADD)=520 padded to multiple of 32

typedef __attribute__((ext_vector_type(8)))  float  v8f;
typedef __attribute__((ext_vector_type(16))) __bf16 v16bf;

__device__ __forceinline__ unsigned short f2bf(float x) {
    unsigned int u = __float_as_uint(x);
    unsigned int r = u + 0x7FFFu + ((u >> 16) & 1u);   // round-to-nearest-even
    return (unsigned short)(r >> 16);
}

__device__ __forceinline__ void fatomic_add(float* p, float v) {
    // no-return agent-scope f32 atomic -> global_atomic_add_f32 (STOREcnt)
    __hip_atomic_fetch_add(p, v, __ATOMIC_RELAXED, __HIP_MEMORY_SCOPE_AGENT);
}

// CDNA5 async copy: 16B global -> LDS, tracked by ASYNCcnt
__device__ __forceinline__ void async_copy_b128(unsigned lds_off, const void* gaddr) {
    asm volatile("global_load_async_to_lds_b128 %0, %1, off"
                 :: "v"(lds_off), "v"(gaddr) : "memory");
}
__device__ __forceinline__ void wait_async0() {
    asm volatile("s_wait_asynccnt 0" ::: "memory");
}

// ---------------------------------------------------------------------------
// Utility kernels
// ---------------------------------------------------------------------------
__global__ void k_zero4(float4* __restrict__ p, size_t n4) {
    size_t i = (size_t)blockIdx.x * blockDim.x + threadIdx.x;
    size_t stride = (size_t)gridDim.x * blockDim.x;
    float4 z; z.x = 0.f; z.y = 0.f; z.z = 0.f; z.w = 0.f;
    for (; i < n4; i += stride) p[i] = z;
}

// W [K][OUT] f32  ->  Wt [OUT][KPAD] bf16 (zero padded K)
__global__ void k_convert_wt(const float* __restrict__ W,
                             unsigned short* __restrict__ Wt,
                             int K, int KPAD, int OUT) {
    size_t total = (size_t)OUT * KPAD;
    size_t i = (size_t)blockIdx.x * blockDim.x + threadIdx.x;
    size_t stride = (size_t)gridDim.x * blockDim.x;
    for (; i < total; i += stride) {
        int o = (int)(i / KPAD);
        int k = (int)(i % KPAD);
        Wt[i] = (k < K) ? f2bf(W[(size_t)k * OUT + o]) : (unsigned short)0;
    }
}

__global__ void k_degrees(const int* __restrict__ src, const int* __restrict__ dst,
                          float* __restrict__ degOut, float* __restrict__ degIn, int nE) {
    int e = blockIdx.x * blockDim.x + threadIdx.x;
    if (e >= nE) return;
    fatomic_add(degOut + src[e], 1.0f);
    fatomic_add(degIn  + dst[e], 1.0f);
}

__global__ void k_rsqrt_clip(float* __restrict__ p, int n) {
    int i = blockIdx.x * blockDim.x + threadIdx.x;
    if (i < n) {
        float d = p[i];
        p[i] = rsqrtf(d > 1.0f ? d : 1.0f);
    }
}

// One wave per edge: agg[dst] += h[src] * rsOut[src]
__global__ __launch_bounds__(256)
void k_aggregate(const int* __restrict__ src, const int* __restrict__ dst,
                 const float* __restrict__ h, const float* __restrict__ rsOut,
                 float* __restrict__ agg, int nE, int dim, int strideIn, int strideOut) {
    int e = blockIdx.x * 8 + (threadIdx.x >> 5);
    if (e >= nE) return;
    int lane = threadIdx.x & 31;
    int s = src[e], d = dst[e];
    float sc = rsOut[s];
    const float* hp = h + (size_t)s * strideIn;
    float* ap = agg + (size_t)d * strideOut;
    if (dim == HDIM) {
        const float4* hp4 = (const float4*)hp;
#pragma unroll
        for (int i = 0; i < HDIM / 128; ++i) {        // 2 x float4 per lane
            int idx = lane + 32 * i;
            float4 v = hp4[idx];
            float* b = ap + idx * 4;
            fatomic_add(b + 0, v.x * sc);
            fatomic_add(b + 1, v.y * sc);
            fatomic_add(b + 2, v.z * sc);
            fatomic_add(b + 3, v.w * sc);
        }
    } else {
        for (int c = lane; c < dim; c += 32)
            fatomic_add(ap + c, hp[c] * sc);
    }
}

// Per-graph readout: z[g, colOff + c] = sum over 40 contiguous rows of h
__global__ __launch_bounds__(256)
void k_readout(const float* __restrict__ h, float* __restrict__ z, int colOff) {
    int g = blockIdx.x;
    int c = threadIdx.x;            // 0..255
    const float* hp = h + (size_t)g * NPG * HDIM + c;
    float s = 0.0f;
#pragma unroll 8
    for (int r = 0; r < NPG; ++r) s += hp[(size_t)r * HDIM];
    z[(size_t)g * ZDIM + colOff + c] = s;
}

// z cols 512..519 = add_features; cols 520..543 = 0
__global__ void k_addfeat(const float* __restrict__ add, float* __restrict__ z) {
    int g = blockIdx.x;
    int j = threadIdx.x;            // 0..31
    float v = (j < ADDF) ? add[(size_t)g * ADDF + j] : 0.0f;
    z[(size_t)g * ZDIM + 2 * HDIM + j] = v;
}

// out[g] = dot(h2[g,:512], w3) + b3   (one wave per graph)
__global__ __launch_bounds__(256)
void k_final(const float* __restrict__ h2, const float* __restrict__ w3,
             const float* __restrict__ b3, float* __restrict__ out) {
    int g = blockIdx.x * 8 + (threadIdx.x >> 5);
    int lane = threadIdx.x & 31;
    const float* hp = h2 + (size_t)g * M2;
    float s = 0.0f;
#pragma unroll
    for (int i = 0; i < M2 / 32; ++i)
        s += hp[lane + 32 * i] * w3[lane + 32 * i];
    for (int o = 16; o > 0; o >>= 1)
        s += __shfl_down(s, o, 32);
    if (lane == 0) out[g] = s + b3[0];
}

// ---------------------------------------------------------------------------
// WMMA fragment helpers (16-bit A/B layout for v_wmma_f32_16x16x32_bf16):
//   pairs v=0..3 -> K = base + half*8 + {0..7}
//   pairs v=4..7 -> K = base + 16 + half*8 + {0..7}
// ---------------------------------------------------------------------------
union frag16 { v16bf v; uint4 q[2]; };

// ---------------------------------------------------------------------------
// Conv GEMM (OUT=256): A staged to LDS as scaled bf16; B panels staged to LDS
// per column tile via CDNA5 async global->LDS copies (ASYNCcnt).
//   Block: 256 threads = 4 row strips x 2 column groups; 64 rows per block.
// ---------------------------------------------------------------------------
template<int KPAD, int ACT>
__global__ __launch_bounds__(256)
void k_gemm_conv(const float* __restrict__ A, const float* __restrict__ rs,
                 const unsigned short* __restrict__ Wt, const float* __restrict__ bias,
                 float* __restrict__ Out) {
    constexpr int RB = 4, CG = 2, OUT = HDIM;
    __shared__ unsigned short sA[RB * 16 * KPAD];
    __shared__ unsigned short sB[CG][16 * KPAD];

    const int rowBase = blockIdx.x * (RB * 16);

    // Stage A: f32 -> bf16 with D_in^{-1/2} row scale
    for (int i = threadIdx.x; i < RB * 16 * KPAD; i += 256) {
        int r = i / KPAD;
        int gr = rowBase + r;
        sA[i] = f2bf(A[(size_t)gr * KPAD + (i - r * KPAD)] * rs[gr]);
    }

    const int lane  = threadIdx.x & 31;
    const int w     = threadIdx.x >> 5;
    const int strip = w & (RB - 1);
    const int cg    = w >> 2;
    const int half  = lane >> 4;
    const int m     = lane & 15;

    const unsigned short* aRow = &sA[(strip * 16 + m) * KPAD];
    constexpr int CHUNKS = KPAD / 8;            // 16-byte chunks per column
    constexpr int TOTAL  = CG * 16 * CHUNKS;

    for (int nt0 = 0; nt0 < OUT / 16 / CG; ++nt0) {
        // Cooperative async stage of CG B panels (16 cols x KPAD bf16 each)
        for (int i = threadIdx.x; i < TOTAL; i += 256) {
            int p   = i / (16 * CHUNKS);
            int rem = i - p * 16 * CHUNKS;
            int n   = rem / CHUNKS;
            int c   = rem - n * CHUNKS;
            const unsigned short* g =
                Wt + (size_t)((nt0 * CG + p) * 16 + n) * KPAD + c * 8;
            unsigned lds = (unsigned)(unsigned long long)&sB[p][n * KPAD + c * 8];
            async_copy_b128(lds, g);
        }
        wait_async0();
        __syncthreads();                         // A + B panels visible

        const int nt  = nt0 * CG + cg;
        const int col = nt * 16 + m;
        const unsigned short* bCol = &sB[cg][m * KPAD];
        v8f acc = {};
#pragma unroll
        for (int kt = 0; kt < KPAD / 32; ++kt) {
            frag16 af, bf;
            const int off0 = kt * 32 + half * 8;
            af.q[0] = *(const uint4*)(aRow + off0);
            af.q[1] = *(const uint4*)(aRow + off0 + 16);
            bf.q[0] = *(const uint4*)(bCol + off0);
            bf.q[1] = *(const uint4*)(bCol + off0 + 16);
            acc = __builtin_amdgcn_wmma_f32_16x16x32_bf16(
                false, af.v, false, bf.v, (short)0, acc, false, false);
        }
        const float bc = bias[col];
#pragma unroll
        for (int r = 0; r < 8; ++r) {
            float v = acc[r] + bc;
            if (ACT == 1) v = v > 0.0f ? v : 0.0f;
            const int grow = rowBase + strip * 16 + half * 8 + r;
            Out[(size_t)grow * OUT + col] = v;
        }
        __syncthreads();                         // before panel overwrite
    }
}

// ---------------------------------------------------------------------------
// Generic WMMA GEMM for the MLP head (B from global; K too large for panels)
//   Block: 256 threads = RB row strips x CG column groups.
//   ACT: 2 = leaky(0.01)
// ---------------------------------------------------------------------------
template<int RB, int CG, int KPAD, int OUT, int ACT>
__global__ __launch_bounds__(256)
void k_gemm(const float* __restrict__ A,
            const unsigned short* __restrict__ Wt, const float* __restrict__ bias,
            float* __restrict__ Out) {
    static_assert(RB * CG == 8, "8 waves per block");
    __shared__ unsigned short sA[RB * 16 * KPAD];

    const int rowBase = blockIdx.x * (RB * 16);
    for (int i = threadIdx.x; i < RB * 16 * KPAD; i += 256) {
        int r = i / KPAD;
        sA[i] = f2bf(A[(size_t)(rowBase + r) * KPAD + (i - r * KPAD)]);
    }
    __syncthreads();

    const int lane  = threadIdx.x & 31;
    const int w     = threadIdx.x >> 5;
    const int strip = w % RB;
    const int cg    = w / RB;
    const int half  = lane >> 4;
    const int m     = lane & 15;

    const unsigned short* aRow = &sA[(strip * 16 + m) * KPAD];

    for (int nt = cg; nt < OUT / 16; nt += CG) {
        const int col = nt * 16 + m;
        const unsigned short* bCol = Wt + (size_t)col * KPAD;
        v8f acc = {};
#pragma unroll
        for (int kt = 0; kt < KPAD / 32; ++kt) {
            frag16 af, bf;
            const int off0 = kt * 32 + half * 8;
            af.q[0] = *(const uint4*)(aRow + off0);
            af.q[1] = *(const uint4*)(aRow + off0 + 16);
            bf.q[0] = *(const uint4*)(bCol + off0);
            bf.q[1] = *(const uint4*)(bCol + off0 + 16);
            acc = __builtin_amdgcn_wmma_f32_16x16x32_bf16(
                false, af.v, false, bf.v, (short)0, acc, false, false);
        }
        const float bc = bias[col];
#pragma unroll
        for (int r = 0; r < 8; ++r) {
            float v = acc[r] + bc;
            if (ACT == 2) v = v > 0.0f ? v : 0.01f * v;
            const int grow = rowBase + strip * 16 + half * 8 + r;
            Out[(size_t)grow * OUT + col] = v;
        }
    }
}

// ---------------------------------------------------------------------------
// Host-side orchestration
// ---------------------------------------------------------------------------
extern "C" void kernel_launch(void* const* d_in, const int* in_sizes, int n_in,
                              void* d_out, int out_size, void* d_ws, size_t ws_size,
                              hipStream_t stream) {
    (void)in_sizes; (void)n_in; (void)out_size; (void)ws_size;

    const int*   cat_src = (const int*)d_in[0];
    const int*   cat_dst = (const int*)d_in[1];
    const float* cat_x   = (const float*)d_in[3];
    const int*   an_src  = (const int*)d_in[4];
    const int*   an_dst  = (const int*)d_in[5];
    const float* an_x    = (const float*)d_in[7];
    const float* addf    = (const float*)d_in[8];
    const float* W0      = (const float*)d_in[9];
    const float* b0      = (const float*)d_in[10];
    const float* gcrW    = (const float*)d_in[11];
    const float* gcrb    = (const float*)d_in[12];
    const float* mW1     = (const float*)d_in[13];
    const float* mb1     = (const float*)d_in[14];
    const float* mW2     = (const float*)d_in[15];
    const float* mb2     = (const float*)d_in[16];
    const float* mW3     = (const float*)d_in[17];
    const float* mb3     = (const float*)d_in[18];
    float* out = (float*)d_out;

    // ---- workspace carving -------------------------------------------------
    char* p = (char*)d_ws;
    auto carve = [&](size_t bytes) -> void* {
        void* r = (void*)p;
        p += (bytes + 255) & ~(size_t)255;
        return r;
    };
    float* rsOut = (float*)carve((size_t)NN * 4);
    float* rsIn  = (float*)carve((size_t)NN * 4);
    float* agg   = (float*)carve((size_t)NN * HDIM * 4);
    float* h     = (float*)carve((size_t)NN * HDIM * 4);
    float* z     = (float*)carve((size_t)NB * ZDIM * 4);
    float* h1    = (float*)carve((size_t)NB * M1 * 4);
    float* h2    = (float*)carve((size_t)NB * M2 * 4);
    unsigned short* Wt0 = (unsigned short*)carve((size_t)HDIM * K1PAD * 2);
    unsigned short* Wtg = (unsigned short*)carve((size_t)10 * HDIM * HDIM * 2);
    unsigned short* Wt1 = (unsigned short*)carve((size_t)M1 * ZDIM * 2);
    unsigned short* Wt2 = (unsigned short*)carve((size_t)M2 * M1 * 2);

    // ---- weight conversion (f32 -> transposed bf16, K-padded) -------------
    {
        int g0 = (HDIM * K1PAD + 255) / 256;
        k_convert_wt<<<g0, 256, 0, stream>>>(W0, Wt0, IN_DIM, K1PAD, HDIM);
        int gg = (HDIM * HDIM + 255) / 256;
        for (int l = 0; l < 2 * GCRL; ++l)
            k_convert_wt<<<gg, 256, 0, stream>>>(gcrW + (size_t)l * HDIM * HDIM,
                                                 Wtg + (size_t)l * HDIM * HDIM,
                                                 HDIM, HDIM, HDIM);
        int g1 = (M1 * ZDIM + 255) / 256;
        k_convert_wt<<<g1, 256, 0, stream>>>(mW1, Wt1, 2 * HDIM + ADDF, ZDIM, M1);
        int g2 = (M2 * M1 + 255) / 256;
        k_convert_wt<<<g2, 256, 0, stream>>>(mW2, Wt2, M1, M1, M2);
    }

    const int gemmGridConv = NN / 64;     // RB=4 -> 64 rows per block
    const int aggGrid = NE / 8;           // one wave per edge, 8 per block
    auto zero = [&](float* ptr, size_t n) {
        size_t n4 = n / 4;
        int blocks = (int)((n4 + 255) / 256);
        if (blocks > 16384) blocks = 16384;
        k_zero4<<<blocks, 256, 0, stream>>>((float4*)ptr, n4);
    };

    // ---- per-side feature extraction --------------------------------------
    for (int side = 0; side < 2; ++side) {
        const int*   src = side ? an_src : cat_src;
        const int*   dst = side ? an_dst : cat_dst;
        const float* x   = side ? an_x   : cat_x;

        // degrees -> rsqrt(clip(deg,1))
        zero(rsOut, NN);
        zero(rsIn,  NN);
        k_degrees<<<(NE + 255) / 256, 256, 0, stream>>>(src, dst, rsOut, rsIn, NE);
        k_rsqrt_clip<<<(NN + 255) / 256, 256, 0, stream>>>(rsOut, NN);
        k_rsqrt_clip<<<(NN + 255) / 256, 256, 0, stream>>>(rsIn,  NN);

        // first conv: x(74) -> h(256), no activation
        zero(agg, (size_t)NN * K1PAD);
        k_aggregate<<<aggGrid, 256, 0, stream>>>(src, dst, x, rsOut, agg,
                                                 NE, IN_DIM, IN_DIM, K1PAD);
        k_gemm_conv<K1PAD, 0><<<gemmGridConv, 256, 0, stream>>>(
            agg, rsIn, Wt0, b0, h);

        // 5 GCR layers, each = conv+relu, conv+relu
        for (int l = 0; l < 2 * GCRL; ++l) {
            zero(agg, (size_t)NN * HDIM);
            k_aggregate<<<aggGrid, 256, 0, stream>>>(src, dst, h, rsOut, agg,
                                                     NE, HDIM, HDIM, HDIM);
            k_gemm_conv<HDIM, 1><<<gemmGridConv, 256, 0, stream>>>(
                agg, rsIn, Wtg + (size_t)l * HDIM * HDIM,
                gcrb + (size_t)l * HDIM, h);
        }

        // readout: per-graph sum into z (cat -> cols 0..255, an -> 256..511)
        k_readout<<<NB, 256, 0, stream>>>(h, z, side * HDIM);
    }

    // ---- MLP head ----------------------------------------------------------
    k_addfeat<<<NB, 32, 0, stream>>>(addf, z);
    k_gemm<2, 4, ZDIM, M1, 2><<<NB / 32, 256, 0, stream>>>(z, Wt1, mb1, h1);
    k_gemm<2, 4, M1, M2, 2><<<NB / 32, 256, 0, stream>>>(h1, Wt2, mb2, h2);
    k_final<<<NB / 8, 256, 0, stream>>>(h2, mW3, mb3, out);
}
